// Block_73976516706525
// MI455X (gfx1250) — matrix operand order repaired and checked by reading, not compiled, not run on hardware
//
#include <hip/hip_runtime.h>
#include <hip/hip_bf16.h>

typedef __attribute__((ext_vector_type(16))) _Float16 v16h;
typedef __attribute__((ext_vector_type(8)))  _Float16 v8h;
typedef __attribute__((ext_vector_type(8)))  float    v8f;
typedef __attribute__((ext_vector_type(4)))  float    v4f;

static constexpr int kB = 2, kS = 1024, kD = 768, kH = 12, kDH = 64, kMLP = 3072;
static constexpr long long kSD = (long long)kS * kD;          // 786432
static constexpr float kScale = 0.125f;                       // 1/sqrt(64)
static constexpr float kThresh = 0.01f;

// ---------------------------------------------------------------------------
// WMMA GEMM: C[M,N] = A[M,K] * op(B) (+bias[n]) (gelu?) (+resid)
// A row-major [M,K]. op(B): transB=0 -> B[K,N] row-major; transB=1 -> B[N,K].
// Block: 256 threads = 8 waves; block tile 128(M) x 64(N); K step 32.
// Wave w owns M rows [w*16, w*16+16); computes 4 WMMAs (N strips of 16).
// LDS: A tile row-major [128][40] halves; B tile TRANSPOSED [n][k] = [64][40].
// Fragment reads are contiguous 16B chunks -> ds_load_b128.
// ---------------------------------------------------------------------------
#define LDA 40
#define LDBT 40

__device__ __forceinline__ float gelu_exact(float x) {
    return 0.5f * x * (1.0f + erff(x * 0.70710678118654752f));
}

__device__ __forceinline__ v8h pack8(v4f a, v4f b) {
    v8h r;
    r[0] = (_Float16)a[0]; r[1] = (_Float16)a[1];
    r[2] = (_Float16)a[2]; r[3] = (_Float16)a[3];
    r[4] = (_Float16)b[0]; r[5] = (_Float16)b[1];
    r[6] = (_Float16)b[2]; r[7] = (_Float16)b[3];
    return r;
}

__global__ __launch_bounds__(256)
void gemm_wmma(const float* __restrict__ A, const float* __restrict__ Bm,
               const float* __restrict__ bias, const float* __restrict__ resid,
               float* __restrict__ C,
               int M, int N, int K,
               long long sA, long long sB, long long sC, long long sR,
               int transB, int doGelu) {
    __shared__ _Float16 sa[128 * LDA];   // A tile, row-major [m][k]
    __shared__ _Float16 sb[64 * LDBT];   // B tile, transposed [n][k]

    const int tid  = threadIdx.x;
    const int lane = tid & 31;
    const int wave = tid >> 5;           // 0..7 = M sub-tile
    const int col  = lane & 15;          // A row within tile / B,C column
    const int kg   = lane >> 4;          // WMMA K/M group selector

    const int gm0 = blockIdx.y * 128;
    const int gn0 = blockIdx.x * 64;
    const long long z = blockIdx.z;

    A  += z * sA;
    Bm += z * sB;
    C  += z * sC;
    const float* R = resid ? resid + z * sR : nullptr;

    // --- cooperative-load assignments (loop-invariant) ---
    // A tile: 128 rows x 32 k; thread -> row ar = tid>>1, k base ac = (tid&1)*16
    const int ar = tid >> 1, ac = (tid & 1) * 16;
    const float* Arow = A + (long long)(gm0 + ar) * K + ac;
    _Float16* sa_w = &sa[ar * LDA + ac];
    // B tile: 64 n x 32 k
    //   transB=1: thread -> n bn1 = tid>>2, k base bk1 = (tid&3)*8 (v4f loads)
    //   transB=0: thread -> n bn0 = tid&63, k base bk0 = (tid>>6)*8 (coalesced b32)
    const int bn1 = tid >> 2, bk1 = (tid & 3) * 8;
    const int bn0 = tid & 63, bk0 = (tid >> 6) * 8;
    const float* Brow1 = Bm + (long long)(gn0 + bn1) * K + bk1;
    const float* Bcol0 = Bm + gn0 + bn0;
    _Float16* sb_w1 = &sb[bn1 * LDBT + bk1];
    _Float16* sb_w0 = &sb[bn0 * LDBT + bk0];

    // --- fragment LDS pointers (loop-invariant) ---
    const _Float16* apf = &sa[(wave * 16 + col) * LDA + kg * 8];   // +16 for hi
    const _Float16* bpf = &sb[col * LDBT + kg * 8];                // + j*16*LDBT

    v8f acc0 = {}, acc1 = {}, acc2 = {}, acc3 = {};

    for (int k0 = 0; k0 < K; k0 += 32) {
        // fill A tile: 16 floats/thread -> two v8h stores
        {
            v4f a0 = *(const v4f*)(Arow + k0);
            v4f a1 = *(const v4f*)(Arow + k0 + 4);
            v4f a2 = *(const v4f*)(Arow + k0 + 8);
            v4f a3 = *(const v4f*)(Arow + k0 + 12);
            *(v8h*)(sa_w)     = pack8(a0, a1);
            *(v8h*)(sa_w + 8) = pack8(a2, a3);
        }
        // fill B tile (transposed layout): 8 halves/thread -> one v8h store
        if (transB) {
            v4f b0 = *(const v4f*)(Brow1 + k0);
            v4f b1 = *(const v4f*)(Brow1 + k0 + 4);
            *(v8h*)(sb_w1) = pack8(b0, b1);
        } else {
            const float* p = Bcol0 + (long long)(k0 + bk0) * N;
            v8h pk;
            #pragma unroll
            for (int j = 0; j < 8; ++j) pk[j] = (_Float16)p[(long long)j * N];
            *(v8h*)(sb_w0) = pk;
        }
        __syncthreads();

        // A fragment: halves [kg*8 .. kg*8+7] and [16+kg*8 .. +7] of this row
        v8h alo = *(const v8h*)(apf);
        v8h ahi = *(const v8h*)(apf + 16);
        v16h af = __builtin_shufflevector(alo, ahi, 0, 1, 2, 3, 4, 5, 6, 7,
                                          8, 9, 10, 11, 12, 13, 14, 15);
        // B fragments: halves [kg*8 .. kg*8+7] and [16+kg*8 .. +7] in transposed tile
        #pragma unroll
        for (int j = 0; j < 4; ++j) {
            const _Float16* bp = bpf + j * 16 * LDBT;
            v8h blo = *(const v8h*)(bp);
            v8h bhi = *(const v8h*)(bp + 16);
            v16h bf = __builtin_shufflevector(blo, bhi, 0, 1, 2, 3, 4, 5, 6, 7,
                                              8, 9, 10, 11, 12, 13, 14, 15);
            if (j == 0) acc0 = __builtin_amdgcn_wmma_f32_16x16x32_f16(false, af, false, bf, (short)0, acc0, false, false);
            if (j == 1) acc1 = __builtin_amdgcn_wmma_f32_16x16x32_f16(false, af, false, bf, (short)0, acc1, false, false);
            if (j == 2) acc2 = __builtin_amdgcn_wmma_f32_16x16x32_f16(false, af, false, bf, (short)0, acc2, false, false);
            if (j == 3) acc3 = __builtin_amdgcn_wmma_f32_16x16x32_f16(false, af, false, bf, (short)0, acc3, false, false);
        }
        __syncthreads();
    }

    // epilogue: C/D layout: VGPR r -> M = r + 8*kg, N = col, per 16-wide strip
    const int mb = gm0 + wave * 16 + kg * 8;
    v8f* accs[4] = { &acc0, &acc1, &acc2, &acc3 };
    #pragma unroll
    for (int j = 0; j < 4; ++j) {
        const int n0 = gn0 + j * 16 + col;
        const float bv = bias ? bias[n0] : 0.f;
        #pragma unroll
        for (int r = 0; r < 8; ++r) {
            long long m = mb + r;
            float v = (*accs[j])[r] + bv;
            if (doGelu) v = gelu_exact(v);
            if (R) v += R[m * N + n0];
            C[m * N + n0] = v;
        }
    }
}

// ---------------------------------------------------------------------------
// DCT matrix generation (orthonormal DCT-II): M[k,i]; optional transpose out.
// ---------------------------------------------------------------------------
__global__ void gen_dct(float* __restrict__ M, float* __restrict__ MT, int n) {
    long long idx = (long long)blockIdx.x * 256 + threadIdx.x;
    if (idx >= (long long)n * n) return;
    int k = (int)(idx / n), i = (int)(idx % n);
    double s = (k == 0) ? sqrt(1.0 / n) : sqrt(2.0 / n);
    float v = (float)(s * cos(3.14159265358979323846 * (2.0 * i + 1.0) * k / (2.0 * n)));
    M[(long long)k * n + i] = v;
    if (MT) MT[(long long)i * n + k] = v;
}

// ---------------------------------------------------------------------------
// LayerNorm over last dim (kD) — one block per row.
// ---------------------------------------------------------------------------
__global__ __launch_bounds__(256)
void ln_kernel(const float* __restrict__ x, const float* __restrict__ g,
               const float* __restrict__ b, float* __restrict__ y) {
    __shared__ float red[256];
    long long row = blockIdx.x;
    const float* xr = x + row * kD;
    float* yr = y + row * kD;
    int t = threadIdx.x;

    float s = 0.f;
    for (int i = t; i < kD; i += 256) s += xr[i];
    red[t] = s; __syncthreads();
    for (int o = 128; o > 0; o >>= 1) { if (t < o) red[t] += red[t + o]; __syncthreads(); }
    float mu = red[0] / kD; __syncthreads();

    float v = 0.f;
    for (int i = t; i < kD; i += 256) { float d = xr[i] - mu; v += d * d; }
    red[t] = v; __syncthreads();
    for (int o = 128; o > 0; o >>= 1) { if (t < o) red[t] += red[t + o]; __syncthreads(); }
    float inv = rsqrtf(red[0] / kD + 1e-6f);

    for (int i = t; i < kD; i += 256) yr[i] = (xr[i] - mu) * inv * g[i] + b[i];
}

// Hard threshold: x *= (|x| > 0.01)
__global__ void thresh_kernel(float* __restrict__ x, long long n) {
    long long i = (long long)blockIdx.x * 256 + threadIdx.x;
    if (i < n) { float v = x[i]; x[i] = (fabsf(v) > kThresh) ? v : 0.f; }
}

// Depthwise 3x3 SAME on [B,D,32,32] view of [B,S,D]; output in [B,S,D].
__global__ void dwconv_kernel(const float* __restrict__ m, const float* __restrict__ w,
                              const float* __restrict__ bias, float* __restrict__ out) {
    long long idx = (long long)blockIdx.x * 256 + threadIdx.x;
    if (idx >= (long long)kB * kS * kD) return;
    int d = (int)(idx % kD);
    long long bs = idx / kD;
    int s = (int)(bs % kS), b = (int)(bs / kS);
    int h = s >> 5, ww = s & 31;
    float acc = bias[d];
    #pragma unroll
    for (int kh = 0; kh < 3; ++kh) {
        int hh = h + kh - 1;
        if (hh < 0 || hh > 31) continue;
        #pragma unroll
        for (int kw = 0; kw < 3; ++kw) {
            int w2 = ww + kw - 1;
            if (w2 < 0 || w2 > 31) continue;
            acc += m[((long long)b * kS + (hh * 32 + w2)) * kD + d] * w[d * 9 + kh * 3 + kw];
        }
    }
    out[idx] = acc;
}

// AvgPool2d(4,4) over (S,DH): [B,S,D] -> [B,H,256,16]
__global__ void pool4_kernel(const float* __restrict__ m, float* __restrict__ out) {
    long long idx = (long long)blockIdx.x * 256 + threadIdx.x;
    if (idx >= (long long)kB * kH * 256 * 16) return;
    int e = (int)(idx % 16);
    int q = (int)((idx / 16) % 256);
    int h = (int)((idx / (16 * 256)) % kH);
    int b = (int)(idx / (16 * 256 * kH));
    float acc = 0.f;
    #pragma unroll
    for (int i = 0; i < 4; ++i)
        #pragma unroll
        for (int j = 0; j < 4; ++j)
            acc += m[((long long)b * kS + (q * 4 + i)) * kD + h * kDH + e * 4 + j];
    out[idx] = acc * (1.f / 16.f);
}

// scores[bh,q,k] = <qp[bh,q,:], kp[bh,k,:]> * scale
__global__ void qk_kernel(const float* __restrict__ qp, const float* __restrict__ kp,
                          float* __restrict__ sc) {
    long long idx = (long long)blockIdx.x * 256 + threadIdx.x;
    if (idx >= (long long)kB * kH * 256 * 256) return;
    int k  = (int)(idx % 256);
    int q  = (int)((idx / 256) % 256);
    long long bh = idx / (256 * 256);
    const float* qr = qp + (bh * 256 + q) * 16;
    const float* kr = kp + (bh * 256 + k) * 16;
    float acc = 0.f;
    #pragma unroll
    for (int e = 0; e < 16; ++e) acc += qr[e] * kr[e];
    sc[idx] = acc * kScale;
}

// softmax over rows of length 256 (one block per row)
__global__ __launch_bounds__(256)
void softmax256_kernel(float* __restrict__ sc) {
    __shared__ float red[256];
    float* r = sc + (long long)blockIdx.x * 256;
    int t = threadIdx.x;
    float v = r[t];
    red[t] = v; __syncthreads();
    for (int o = 128; o > 0; o >>= 1) { if (t < o) red[t] = fmaxf(red[t], red[t + o]); __syncthreads(); }
    float mx = red[0]; __syncthreads();
    float e = expf(v - mx);
    red[t] = e; __syncthreads();
    for (int o = 128; o > 0; o >>= 1) { if (t < o) red[t] += red[t + o]; __syncthreads(); }
    r[t] = e / red[0];
}

// cont[bh,q,e] = sum_k att[bh,q,k] * vp[bh,k,e]
__global__ void attnv_kernel(const float* __restrict__ sc, const float* __restrict__ vp,
                             float* __restrict__ cs) {
    long long idx = (long long)blockIdx.x * 256 + threadIdx.x;
    if (idx >= (long long)kB * kH * 256 * 16) return;
    int e = (int)(idx % 16);
    int q = (int)((idx / 16) % 256);
    long long bh = idx / (16 * 256);
    const float* ar = sc + (bh * 256 + q) * 256;
    const float* vb = vp + bh * 256 * 16 + e;
    float acc = 0.f;
    for (int k = 0; k < 256; ++k) acc += ar[k] * vb[k * 16];
    cs[idx] = acc;
}

// bilinear x4 upsample (half-pixel): [B,H,256,16] -> [B,H,S,64]
__global__ void upsample_kernel(const float* __restrict__ cs, float* __restrict__ cup) {
    long long idx = (long long)blockIdx.x * 256 + threadIdx.x;
    if (idx >= (long long)kB * kH * kS * kDH) return;
    int e = (int)(idx % kDH);
    int s = (int)((idx / kDH) % kS);
    long long bh = idx / ((long long)kDH * kS);
    float sf = (s + 0.5f) * 0.25f - 0.5f;
    float ef = (e + 0.5f) * 0.25f - 0.5f;
    int s0 = (int)floorf(sf); float ws = sf - (float)s0;
    int e0 = (int)floorf(ef); float we = ef - (float)e0;
    int s0c = min(max(s0, 0), 255), s1c = min(max(s0 + 1, 0), 255);
    int e0c = min(max(e0, 0), 15),  e1c = min(max(e0 + 1, 0), 15);
    const float* base = cs + bh * 256 * 16;
    float v00 = base[s0c * 16 + e0c], v01 = base[s0c * 16 + e1c];
    float v10 = base[s1c * 16 + e0c], v11 = base[s1c * 16 + e1c];
    cup[idx] = (1.f - ws) * ((1.f - we) * v00 + we * v01)
             +        ws  * ((1.f - we) * v10 + we * v11);
}

// elementwise "attention": softmax over DH of q2*k2*scale, times v2. wave32/row.
__global__ __launch_bounds__(128)
void ewattn_kernel(const float* __restrict__ q2, const float* __restrict__ k2,
                   const float* __restrict__ v2, float* __restrict__ ctx) {
    int lane = threadIdx.x & 31;
    int wv   = threadIdx.x >> 5;
    long long row = (long long)blockIdx.x * 4 + wv;          // over B*S*H
    if (row >= (long long)kB * kS * kH) return;
    int h = (int)(row % kH);
    long long bs = row / kH;
    long long base = bs * kD + h * kDH;
    float p0 = q2[base + lane]      * k2[base + lane]      * kScale;
    float p1 = q2[base + lane + 32] * k2[base + lane + 32] * kScale;
    float mx = fmaxf(p0, p1);
    for (int o = 16; o > 0; o >>= 1) mx = fmaxf(mx, __shfl_xor(mx, o, 32));
    float e0 = expf(p0 - mx), e1 = expf(p1 - mx);
    float s = e0 + e1;
    for (int o = 16; o > 0; o >>= 1) s += __shfl_xor(s, o, 32);
    float inv = 1.f / s;
    ctx[base + lane]      = e0 * inv * v2[base + lane];
    ctx[base + lane + 32] = e1 * inv * v2[base + lane + 32];
}

// fuse: out[b,s,o*64+e] = sum_c fuse_w[o,c]*cat[b,c,s,e] + fuse_b[o]
__global__ void fuse_kernel(const float* __restrict__ ctx, const float* __restrict__ cup,
                            const float* __restrict__ fw, const float* __restrict__ fb,
                            float* __restrict__ out) {
    long long idx = (long long)blockIdx.x * 256 + threadIdx.x;
    if (idx >= (long long)kB * kS * kD) return;
    int d = (int)(idx % kD);
    long long bs = idx / kD;
    int s = (int)(bs % kS), b = (int)(bs / kS);
    int o = d >> 6, e = d & 63;
    float acc = fb[o];
    #pragma unroll
    for (int c = 0; c < kH; ++c)
        acc += fw[o * (2 * kH) + c] * ctx[bs * kD + c * kDH + e];
    #pragma unroll
    for (int c = 0; c < kH; ++c)
        acc += fw[o * (2 * kH) + kH + c] * cup[(((long long)b * kH + c) * kS + s) * kDH + e];
    out[idx] = acc;
}

// ---------------------------------------------------------------------------
extern "C" void kernel_launch(void* const* d_in, const int* in_sizes, int n_in,
                              void* d_out, int out_size, void* d_ws, size_t ws_size,
                              hipStream_t stream) {
    const float* x      = (const float*)d_in[0];
    const float* ln1_g  = (const float*)d_in[1];
    const float* ln1_b  = (const float*)d_in[2];
    const float* wq     = (const float*)d_in[3];
    const float* bq     = (const float*)d_in[4];
    const float* wk     = (const float*)d_in[5];
    const float* bk     = (const float*)d_in[6];
    const float* wv     = (const float*)d_in[7];
    const float* bv     = (const float*)d_in[8];
    const float* dw_w   = (const float*)d_in[9];
    const float* dw_b   = (const float*)d_in[10];
    const float* pw_w   = (const float*)d_in[11];
    const float* pw_b   = (const float*)d_in[12];
    const float* fuse_w = (const float*)d_in[13];
    const float* fuse_b = (const float*)d_in[14];
    const float* wo     = (const float*)d_in[15];
    const float* bo     = (const float*)d_in[16];
    const float* ln2_g  = (const float*)d_in[17];
    const float* ln2_b  = (const float*)d_in[18];
    const float* fc1_w  = (const float*)d_in[19];
    const float* fc1_b  = (const float*)d_in[20];
    const float* fc2_w  = (const float*)d_in[21];
    const float* fc2_b  = (const float*)d_in[22];
    float* out = (float*)d_out;

    // ---- workspace layout (floats), with aliasing ----
    float* ws = (float*)d_ws;
    const long long SD = kSD;
    float* DsM  = ws;                 ws += (long long)kS * kS;
    float* DsT  = ws;                 ws += (long long)kS * kS;
    float* DdM  = ws;                 ws += (long long)kD * kD;
    float* XN   = ws;                 ws += kB * SD;  // xn / ctxew / xm
    float* T    = ws;                 ws += kB * SD;  // t1 / scores / t2
    float* XD   = ws;                 ws += kB * SD;  // xd / cup
    float* MQ   = ws;                 ws += kB * SD;  // mq / fused
    float* MK   = ws;                 ws += kB * SD;  // mk / ao
    float* MV   = ws;                 ws += kB * SD;  // mv / x2
    float* Q2   = ws;                 ws += kB * SD;
    float* K2   = ws;                 ws += kB * SD;
    float* V2   = ws;                 ws += kB * SD;
    float* DWT  = ws;                 ws += kB * SD;  // also tail of H1
    float* QP   = ws;                 ws += (long long)kB * kH * 256 * 16;
    float* KP   = ws;                 ws += (long long)kB * kH * 256 * 16;
    float* VP   = ws;                 ws += (long long)kB * kH * 256 * 16;
    float* CS   = ws;                 ws += (long long)kB * kH * 256 * 16;
    float* SCOR = T;                  // 2*SD == B*H*256*256 exactly
    float* CUP  = XD;
    float* CTXE = XN;
    float* FUSD = MQ;
    float* AO   = MK;
    float* X2   = MV;
    float* XM   = XN;
    float* H1   = Q2;                 // Q2..DWT contiguous = B*S*MLP exactly

    const long long nBSD = (long long)kB * kS * kD;
    auto eb = [](long long n) { return (unsigned)((n + 255) / 256); };
    const long long SM = (long long)kS * kMLP;

    // DCT matrices
    gen_dct<<<eb((long long)kS * kS), 256, 0, stream>>>(DsM, DsT, kS);
    gen_dct<<<eb((long long)kD * kD), 256, 0, stream>>>(DdM, nullptr, kD);

    // LN1
    ln_kernel<<<kB * kS, 256, 0, stream>>>(x, ln1_g, ln1_b, XN);

    dim3 blk(256);
    dim3 gNd(kD / 64, kS / 128, kB);     // N=768,  M=1024
    dim3 gNm(kMLP / 64, kS / 128, kB);   // N=3072, M=1024

    // 2-D DCT: t1 = Ds @ xn ; xd = t1 @ Dd^T ; threshold
    gemm_wmma<<<gNd, blk, 0, stream>>>(DsM, XN, nullptr, nullptr, T,
                                       kS, kD, kS, 0, SD, SD, 0, 0, 0);
    gemm_wmma<<<gNd, blk, 0, stream>>>(T, DdM, nullptr, nullptr, XD,
                                       kS, kD, kD, SD, 0, SD, 0, 1, 0);
    thresh_kernel<<<eb(nBSD), 256, 0, stream>>>(XD, nBSD);

    // QKV projections (B transposed = weight [out,in])
    gemm_wmma<<<gNd, blk, 0, stream>>>(XD, wq, bq, nullptr, MQ, kS, kD, kD, SD, 0, SD, 0, 1, 0);
    gemm_wmma<<<gNd, blk, 0, stream>>>(XD, wk, bk, nullptr, MK, kS, kD, kD, SD, 0, SD, 0, 1, 0);
    gemm_wmma<<<gNd, blk, 0, stream>>>(XD, wv, bv, nullptr, MV, kS, kD, kD, SD, 0, SD, 0, 1, 0);

    // Pooled-attention path (small)
    pool4_kernel<<<eb((long long)kB * kH * 256 * 16), 256, 0, stream>>>(MQ, QP);
    pool4_kernel<<<eb((long long)kB * kH * 256 * 16), 256, 0, stream>>>(MK, KP);
    pool4_kernel<<<eb((long long)kB * kH * 256 * 16), 256, 0, stream>>>(MV, VP);

    // Depthwise + pointwise path (q2/k2/v2 in [B,S,D])
    const float* src[3] = { MQ, MK, MV };
    float* dst[3] = { Q2, K2, V2 };
    for (int i = 0; i < 3; ++i) {
        dwconv_kernel<<<eb(nBSD), 256, 0, stream>>>(src[i], dw_w, dw_b, DWT);
        gemm_wmma<<<gNd, blk, 0, stream>>>(DWT, pw_w, pw_b, nullptr, dst[i],
                                           kS, kD, kD, SD, 0, SD, 0, 1, 0);
    }

    // small attention: scores -> softmax -> context -> upsample (uses T as scores)
    qk_kernel<<<eb((long long)kB * kH * 256 * 256), 256, 0, stream>>>(QP, KP, SCOR);
    softmax256_kernel<<<kB * kH * 256, 256, 0, stream>>>(SCOR);
    attnv_kernel<<<eb((long long)kB * kH * 256 * 16), 256, 0, stream>>>(SCOR, VP, CS);
    upsample_kernel<<<eb(nBSD), 256, 0, stream>>>(CS, CUP);

    // elementwise attention on q2/k2/v2 -> ctx (XN buffer)
    ewattn_kernel<<<(unsigned)(((long long)kB * kS * kH + 3) / 4), 128, 0, stream>>>(Q2, K2, V2, CTXE);

    // fuse heads -> FUSD (MQ buffer), then output projection
    fuse_kernel<<<eb(nBSD), 256, 0, stream>>>(CTXE, CUP, fuse_w, fuse_b, FUSD);
    gemm_wmma<<<gNd, blk, 0, stream>>>(FUSD, wo, bo, nullptr, AO, kS, kD, kD, SD, 0, SD, 0, 1, 0);

    // inverse 2-D DCT + residual: t2 = Ds^T @ ao ; x2 = t2 @ Dd + x
    gemm_wmma<<<gNd, blk, 0, stream>>>(DsT, AO, nullptr, nullptr, T,
                                       kS, kD, kS, 0, SD, SD, 0, 0, 0);
    gemm_wmma<<<gNd, blk, 0, stream>>>(T, DdM, nullptr, x, X2,
                                       kS, kD, kD, SD, 0, SD, SD, 0, 0);

    // LN2 + MLP (gelu in fc1 epilogue, residual in fc2 epilogue) -> d_out
    ln_kernel<<<kB * kS, 256, 0, stream>>>(X2, ln2_g, ln2_b, XM);
    gemm_wmma<<<gNm, blk, 0, stream>>>(XM, fc1_w, fc1_b, nullptr, H1,
                                       kS, kMLP, kD, SD, 0, SM, 0, 1, 1);
    gemm_wmma<<<gNd, blk, 0, stream>>>(H1, fc2_w, fc2_b, X2, out,
                                       kS, kD, kMLP, SM, 0, SD, SD, 1, 0);
}
